// EfficientMultiScaleDilatedAttention_4466765987897
// MI455X (gfx1250) — matrix operand order repaired and checked by reading, not compiled, not run on hardware
//
#include <hip/hip_runtime.h>

// ---------------------------------------------------------------------------
// EfficientMultiScaleDilatedAttention for MI455X (gfx1250), bf16 WMMA pipeline
// B=32, N=1024, C=768, H=12, D=64, 128 keys/head (64 linformer + 64 bank)
// Async global->LDS staging (ASYNCcnt) + double-buffered WMMA GEMM.
// ---------------------------------------------------------------------------

typedef __bf16 bf16_t;
typedef __attribute__((ext_vector_type(16))) __bf16 v16bf;
typedef __attribute__((ext_vector_type(8)))  __bf16 v8bf;
typedef __attribute__((ext_vector_type(8)))  float  v8f;
typedef int gv4i __attribute__((vector_size(16)));   // matches builtin param type

#define DEV static __device__ __forceinline__
#define GLOBAL_AS __attribute__((address_space(1)))
#define LDS_AS    __attribute__((address_space(3)))

#if defined(__gfx1250__) && __has_builtin(__builtin_amdgcn_global_load_async_to_lds_b128)
#define HAVE_ASYNC_LDS 1
#else
#define HAVE_ASYNC_LDS 0
#endif

// 16-byte global -> LDS copy (async on gfx1250; sync fallback otherwise).
DEV void cp16(bf16_t* lds, const bf16_t* g) {
#if HAVE_ASYNC_LDS
  __builtin_amdgcn_global_load_async_to_lds_b128(
      (GLOBAL_AS gv4i*)g, (LDS_AS gv4i*)lds, 0, 0);
#else
  *(uint4*)lds = *(const uint4*)g;
#endif
}

DEV void cp_wait() {
#if HAVE_ASYNC_LDS
#if __has_builtin(__builtin_amdgcn_s_wait_asynccnt)
  __builtin_amdgcn_s_wait_asynccnt(0);
#else
  asm volatile("s_wait_asynccnt 0x0" ::: "memory");
#endif
#endif
}

// A-matrix fragment (16x32, M x K), bf16, from row-major [row][k] buffer.
// Lane L holds row = row0 + (L&15); K elems [kk..kk+7, 16+kk..16+kk+7], kk=8*(L>=16).
DEV v16bf ld_frag_a(const bf16_t* base, int stride, int row0, int k0, int lane) {
  int r  = row0 + (lane & 15);
  int kk = (lane < 16) ? 0 : 8;
  const bf16_t* p = base + r * stride + k0 + kk;
  union { v16bf v; v8bf h[2]; } u;
  u.h[0] = *(const v8bf*)p;
  u.h[1] = *(const v8bf*)(p + 16);
  return u.v;
}

// B-matrix fragment (32x16, K x N), bf16, from "Bt" row-major [n][k] buffer.
// Lane L holds col n = n0 + (L&15); K elems [kk..kk+7, 16+kk..16+kk+7], kk=8*(L>=16).
DEV v16bf ld_frag_b(const bf16_t* base, int stride, int n0, int k0, int lane) {
  int r  = n0 + (lane & 15);
  int kk = (lane < 16) ? 0 : 8;
  const bf16_t* p = base + r * stride + k0 + kk;
  union { v16bf v; v8bf h[2]; } u;
  u.h[0] = *(const v8bf*)p;
  u.h[1] = *(const v8bf*)(p + 16);
  return u.v;
}

DEV v8f wmma_bf16(v16bf a, v16bf b, v8f c) {
  return __builtin_amdgcn_wmma_f32_16x16x32_bf16(false, a, false, b, (short)0, c,
                                                 false, false);
}

DEV v8f vzero8() { v8f z = {0.f,0.f,0.f,0.f,0.f,0.f,0.f,0.f}; return z; }

// ---------------------------------------------------------------------------
// Prep kernels
// ---------------------------------------------------------------------------

__global__ void k_cvt(const float* __restrict__ in, bf16_t* __restrict__ out, int n4) {
  int i = blockIdx.x * blockDim.x + threadIdx.x;
  if (i >= n4) return;
  float4 v = ((const float4*)in)[i];
  bf16_t* o = out + (size_t)i * 4;
  o[0] = (bf16_t)v.x; o[1] = (bf16_t)v.y; o[2] = (bf16_t)v.z; o[3] = (bf16_t)v.w;
}

// Pairwise-pool the first 256 tokens of each batch -> 128 tokens (bf16).
__global__ void k_pool(const float* __restrict__ x, bf16_t* __restrict__ xpb) {
  int i = blockIdx.x * 256 + threadIdx.x;            // < 4096*768
  int r = i / 768, c = i % 768;
  int b = r >> 7, q = r & 127;
  size_t base = ((size_t)b * 1024 + 2 * q) * 768 + c;
  xpb[i] = (bf16_t)(0.5f * (x[base] + x[base + 768]));
}

// out[n*768 + c] = bf16(w[c*ldin + col0 + n])  (weight slice transpose -> N x K)
__global__ void k_wt(const float* __restrict__ w, bf16_t* __restrict__ out,
                     int ldin, int col0) {
  int i = blockIdx.x * 256 + threadIdx.x;            // < 768*768
  int nn = i / 768, c = i % 768;
  out[i] = (bf16_t)(w[(size_t)c * ldin + col0 + nn]);
}

// ---------------------------------------------------------------------------
// Generic bf16 WMMA GEMM:  out[M][N] = A[M][K] * Bt[N][K]^T + bias[N]
// Tile 128x128, 8 waves (2 M-tiles x 4 N-tiles each), BK=32.
// Double-buffered LDS, filled by async global->LDS loads overlapping WMMA.
// ---------------------------------------------------------------------------
template <bool OUT_BF16>
__global__ __launch_bounds__(256) void k_gemm(const bf16_t* __restrict__ A,
                                              const bf16_t* __restrict__ Bt,
                                              const float* __restrict__ bias,
                                              void* __restrict__ out,
                                              int M, int Nn, int K) {
  __shared__ __align__(16) bf16_t sA[2][128 * 40];
  __shared__ __align__(16) bf16_t sB[2][128 * 40];

  const int tid  = threadIdx.x, lane = tid & 31, w = tid >> 5;
  const int hf   = lane >> 4,   ln   = lane & 15;
  const int mbase = (w & 3) * 32, nbase = (w >> 2) * 64;
  const int brow = blockIdx.y * 128, bcol = blockIdx.x * 128;

  v8f acc[2][4];
  #pragma unroll
  for (int i = 0; i < 2; i++)
    #pragma unroll
    for (int j = 0; j < 4; j++) acc[i][j] = vzero8();

  const int row = tid >> 1, seg = tid & 1;
  const bf16_t* gA = A  + (size_t)(brow + row) * K + seg * 16;
  const bf16_t* gB = Bt + (size_t)(bcol + row) * K + seg * 16;
  const int soff = row * 40 + seg * 16;

  auto stage = [&](int buf, int k0) {
    cp16(&sA[buf][soff],     gA + k0);
    cp16(&sA[buf][soff + 8], gA + k0 + 8);
    cp16(&sB[buf][soff],     gB + k0);
    cp16(&sB[buf][soff + 8], gB + k0 + 8);
  };

  stage(0, 0);  // prologue fill of buffer 0

  for (int k0 = 0; k0 < K; k0 += 32) {
    const int cur = (k0 >> 5) & 1;
    cp_wait();        // current buffer's async fills complete (this wave)
    __syncthreads();  // all waves filled + all done computing on the other buf
    if (k0 + 32 < K) stage(cur ^ 1, k0 + 32);  // overlaps with WMMA below

    const bf16_t* cA = sA[cur];
    const bf16_t* cB = sB[cur];
    v16bf a0 = ld_frag_a(cA, 40, mbase,      0, lane);
    v16bf a1 = ld_frag_a(cA, 40, mbase + 16, 0, lane);
    #pragma unroll
    for (int nt = 0; nt < 4; nt++) {
      v16bf bb = ld_frag_b(cB, 40, nbase + nt * 16, 0, lane);
      acc[0][nt] = wmma_bf16(a0, bb, acc[0][nt]);
      acc[1][nt] = wmma_bf16(a1, bb, acc[1][nt]);
    }
  }

  #pragma unroll
  for (int mt = 0; mt < 2; mt++)
    #pragma unroll
    for (int nt = 0; nt < 4; nt++) {
      int gcol = bcol + nbase + nt * 16 + ln;
      float bv = bias[gcol];
      #pragma unroll
      for (int r = 0; r < 8; r++) {
        int grow = brow + mbase + mt * 16 + r + 8 * hf;
        float v = acc[mt][nt][r] + bv;
        if (OUT_BF16) ((bf16_t*)out)[(size_t)grow * Nn + gcol] = (bf16_t)v;
        else          ((float*) out)[(size_t)grow * Nn + gcol] = v;
      }
    }
}

// ---------------------------------------------------------------------------
// Linformer compression + bank packing (small, f32):
//   kc[c,d] = sum_s Ek[s,c] * klin[b,s,h*64+d]   (keys 0..63)
//   keys 64..127 from k_bank.  kf: (b,h,key,d); vfT: (b,h,d,key)
// ---------------------------------------------------------------------------
__global__ __launch_bounds__(256) void k_compress(const float* __restrict__ klin,
                                                  const float* __restrict__ vlin,
                                                  const float* __restrict__ Ek,
                                                  const float* __restrict__ Ev,
                                                  const float* __restrict__ kbank,
                                                  const float* __restrict__ vbank,
                                                  bf16_t* __restrict__ kf,
                                                  bf16_t* __restrict__ vfT) {
  __shared__ float sT[128 * 64];
  const int h = blockIdx.x, b = blockIdx.y, tid = threadIdx.x;
  const size_t obase = ((size_t)(b * 12 + h)) * (128 * 64);

  // ---- K ----
  for (int i = tid; i < 128 * 64; i += 256) {
    int s = i >> 6, d = i & 63;
    sT[i] = klin[((size_t)b * 128 + s) * 768 + h * 64 + d];
  }
  __syncthreads();
  for (int i = tid; i < 64 * 64; i += 256) {
    int c = i >> 6, d = i & 63;
    float acc = 0.f;
    for (int s = 0; s < 128; s++) acc += Ek[s * 64 + c] * sT[s * 64 + d];
    kf[obase + (size_t)c * 64 + d] = (bf16_t)acc;
  }
  for (int i = tid; i < 64 * 64; i += 256) {
    int j = i >> 6, d = i & 63;
    kf[obase + (size_t)(64 + j) * 64 + d] =
        (bf16_t)kbank[((size_t)b * 64 + j) * 768 + h * 64 + d];
  }
  __syncthreads();

  // ---- V ----
  for (int i = tid; i < 128 * 64; i += 256) {
    int s = i >> 6, d = i & 63;
    sT[i] = vlin[((size_t)b * 128 + s) * 768 + h * 64 + d];
  }
  __syncthreads();
  for (int i = tid; i < 64 * 64; i += 256) {
    int c = i >> 6, d = i & 63;
    float acc = 0.f;
    for (int s = 0; s < 128; s++) acc += Ev[s * 64 + c] * sT[s * 64 + d];
    vfT[obase + (size_t)d * 128 + c] = (bf16_t)acc;
  }
  for (int i = tid; i < 64 * 64; i += 256) {
    int j = i >> 6, d = i & 63;
    vfT[obase + (size_t)d * 128 + 64 + j] =
        (bf16_t)vbank[((size_t)b * 64 + j) * 768 + h * 64 + d];
  }
}

// ---------------------------------------------------------------------------
// Fused attention: per block = 128 queries of one (b,h).
// scores (128x128) live in WMMA accumulators; softmax via shfl_xor; probs
// bounced through LDS (aliasing dead Q tile) into A-fragments for P@V.
// Q/K/V tiles staged with async global->LDS loads.
// ---------------------------------------------------------------------------
__global__ __launch_bounds__(128) void k_attn(const bf16_t* __restrict__ Qb,
                                              const bf16_t* __restrict__ kf,
                                              const bf16_t* __restrict__ vfT,
                                              bf16_t* __restrict__ ao) {
  __shared__ __align__(16) bf16_t reg0[128 * 136];  // sA (128x72) then sP (128x136)
  __shared__ __align__(16) bf16_t reg1[128 * 72];   // sB (128x72) then sV (64x136)
  bf16_t* sA = reg0;
  bf16_t* sP = reg0;
  bf16_t* sB = reg1;
  bf16_t* sV = reg1;

  const int qt = blockIdx.x, h = blockIdx.y, b = blockIdx.z;
  const int tid = threadIdx.x, lane = tid & 31, w = tid >> 5;
  const int hf = lane >> 4, ln = lane & 15;
  const int q0 = qt * 128;
  const float scale = 0.125f;  // 1/sqrt(64)

  // async-stage Q tile (128 x 64) and K tile (128 keys x 64)
  {
    const bf16_t* src = Qb + ((size_t)(b * 1024 + q0 + tid)) * 768 + h * 64;
    bf16_t* dst = sA + tid * 72;
    #pragma unroll
    for (int j = 0; j < 8; j++) cp16(dst + j * 8, src + j * 8);
  }
  {
    const bf16_t* src = kf + ((size_t)(b * 12 + h)) * (128 * 64) + tid * 64;
    bf16_t* dst = sB + tid * 72;
    #pragma unroll
    for (int j = 0; j < 8; j++) cp16(dst + j * 8, src + j * 8);
  }
  cp_wait();
  __syncthreads();

  // scores: Q(128x64) @ K^T -> wave w owns rows [32w, 32w+32)
  v8f acc[2][8];
  #pragma unroll
  for (int i = 0; i < 2; i++)
    #pragma unroll
    for (int j = 0; j < 8; j++) acc[i][j] = vzero8();
  {
    v16bf a00 = ld_frag_a(sA, 72, (2 * w) * 16,     0,  lane);
    v16bf a01 = ld_frag_a(sA, 72, (2 * w) * 16,     32, lane);
    v16bf a10 = ld_frag_a(sA, 72, (2 * w + 1) * 16, 0,  lane);
    v16bf a11 = ld_frag_a(sA, 72, (2 * w + 1) * 16, 32, lane);
    #pragma unroll
    for (int nt = 0; nt < 8; nt++) {
      v16bf b0 = ld_frag_b(sB, 72, nt * 16, 0,  lane);
      v16bf b1 = ld_frag_b(sB, 72, nt * 16, 32, lane);
      acc[0][nt] = wmma_bf16(a00, b0, acc[0][nt]);
      acc[0][nt] = wmma_bf16(a01, b1, acc[0][nt]);
      acc[1][nt] = wmma_bf16(a10, b0, acc[1][nt]);
      acc[1][nt] = wmma_bf16(a11, b1, acc[1][nt]);
    }
  }
  __syncthreads();  // sA/sB reads complete; regions may be recycled

  // async-stage V tile (64 d-rows x 128 keys) into region1; overlaps softmax
  {
    const bf16_t* src = vfT + ((size_t)(b * 12 + h)) * (64 * 128) +
                        (tid >> 1) * 128 + (tid & 1) * 64;
    bf16_t* dst = sV + (tid >> 1) * 136 + (tid & 1) * 64;
    #pragma unroll
    for (int j = 0; j < 8; j++) cp16(dst + j * 8, src + j * 8);
  }

  // softmax per row (row m = (2w+mt)*16 + r + 8*hf, cols nt*16 + ln)
  #pragma unroll
  for (int mt = 0; mt < 2; mt++) {
    #pragma unroll
    for (int r = 0; r < 8; r++) {
      float pm = -3.0e38f;
      #pragma unroll
      for (int nt = 0; nt < 8; nt++) pm = fmaxf(pm, acc[mt][nt][r] * scale);
      #pragma unroll
      for (int s = 1; s <= 8; s <<= 1) pm = fmaxf(pm, __shfl_xor(pm, s, 32));
      float e[8]; float ps = 0.f;
      #pragma unroll
      for (int nt = 0; nt < 8; nt++) {
        e[nt] = __expf(acc[mt][nt][r] * scale - pm);
        ps += e[nt];
      }
      #pragma unroll
      for (int s = 1; s <= 8; s <<= 1) ps += __shfl_xor(ps, s, 32);
      float inv = 1.0f / ps;
      int m = (2 * w + mt) * 16 + r + 8 * hf;
      #pragma unroll
      for (int nt = 0; nt < 8; nt++)
        sP[m * 136 + nt * 16 + ln] = (bf16_t)(e[nt] * inv);
    }
  }
  cp_wait();        // V tile async fills complete
  __syncthreads();

  // out = P(128x128) @ V(128keys x 64d); Bt layout = sV (d-rows, key-cols)
  #pragma unroll
  for (int mtl = 0; mtl < 2; mtl++) {
    int m0 = (2 * w + mtl) * 16;
    v8f oacc[4];
    #pragma unroll
    for (int nt = 0; nt < 4; nt++) oacc[nt] = vzero8();
    #pragma unroll
    for (int ks = 0; ks < 4; ks++) {
      v16bf a = ld_frag_a(sP, 136, m0, ks * 32, lane);
      #pragma unroll
      for (int nt = 0; nt < 4; nt++) {
        v16bf bb = ld_frag_b(sV, 136, nt * 16, ks * 32, lane);
        oacc[nt] = wmma_bf16(a, bb, oacc[nt]);
      }
    }
    #pragma unroll
    for (int nt = 0; nt < 4; nt++)
      #pragma unroll
      for (int r = 0; r < 8; r++) {
        int m = m0 + r + 8 * hf;
        ao[((size_t)(b * 1024 + q0 + m)) * 768 + h * 64 + nt * 16 + ln] =
            (bf16_t)oacc[nt][r];
      }
  }
}

// ---------------------------------------------------------------------------
// Host launch
// ---------------------------------------------------------------------------
extern "C" void kernel_launch(void* const* d_in, const int* in_sizes, int n_in,
                              void* d_out, int out_size, void* d_ws, size_t ws_size,
                              hipStream_t stream) {
  const float* x     = (const float*)d_in[0];  // (32,1024,768)
  const float* Wqkv  = (const float*)d_in[1];  // (768,2304)
  const float* bqkv  = (const float*)d_in[2];  // (2304,)
  const float* Ek    = (const float*)d_in[3];  // (128,64)
  const float* Ev    = (const float*)d_in[4];  // (128,64)
  const float* Wproj = (const float*)d_in[5];  // (768,768)
  const float* bproj = (const float*)d_in[6];  // (768,)
  const float* kbank = (const float*)d_in[7];  // (32,64,768)
  const float* vbank = (const float*)d_in[8];  // (32,64,768)

  char* ws = (char*)d_ws;
  size_t off = 0;
  auto alloc = [&](size_t bytes) -> void* {
    void* p = ws + off;
    off += (bytes + 255) & ~(size_t)255;
    return p;
  };

  bf16_t* xb  = (bf16_t*)alloc(50331648);  // x in bf16; reused as attention output
  bf16_t* ao  = xb;
  bf16_t* xpb = (bf16_t*)alloc(6291456);   // pooled 128 tokens/batch, bf16
  bf16_t* WqT = (bf16_t*)alloc(1179648);
  bf16_t* WkT = (bf16_t*)alloc(1179648);
  bf16_t* WvT = (bf16_t*)alloc(1179648);
  bf16_t* WpT = (bf16_t*)alloc(1179648);
  bf16_t* Qb  = (bf16_t*)alloc(50331648);  // Q (B*N, C) bf16
  float*  klin = (float*)alloc(12582912);  // (B*128, C) f32
  float*  vlin = (float*)alloc(12582912);
  bf16_t* kfb = (bf16_t*)alloc(6291456);   // k_full (b,h,128,64) bf16
  bf16_t* vft = (bf16_t*)alloc(6291456);   // v_full^T (b,h,64,128) bf16
  (void)ws_size; (void)in_sizes; (void)n_in; (void)out_size;

  // prep
  k_cvt <<<24576, 256, 0, stream>>>(x, xb, 6291456);   // 32*1024*768/4
  k_pool<<<12288, 256, 0, stream>>>(x, xpb);           // 4096*768/256
  k_wt  <<<2304, 256, 0, stream>>>(Wqkv,  WqT, 2304, 0);
  k_wt  <<<2304, 256, 0, stream>>>(Wqkv,  WkT, 2304, 768);
  k_wt  <<<2304, 256, 0, stream>>>(Wqkv,  WvT, 2304, 1536);
  k_wt  <<<2304, 256, 0, stream>>>(Wproj, WpT, 768,  0);

  // projections
  k_gemm<true ><<<dim3(6, 256), 256, 0, stream>>>(xb,  WqT, bqkv,        Qb,   32768, 768, 768);
  k_gemm<false><<<dim3(6, 32),  256, 0, stream>>>(xpb, WkT, bqkv + 768,  klin, 4096,  768, 768);
  k_gemm<false><<<dim3(6, 32),  256, 0, stream>>>(xpb, WvT, bqkv + 1536, vlin, 4096,  768, 768);

  // linformer compression + bank packing
  k_compress<<<dim3(12, 32), 256, 0, stream>>>(klin, vlin, Ek, Ev, kbank, vbank, kfb, vft);

  // fused attention (1024 queries x 128 keys per head)
  k_attn<<<dim3(8, 12, 32), 128, 0, stream>>>(Qb, kfb, vft, ao);

  // output projection
  k_gemm<false><<<dim3(6, 256), 256, 0, stream>>>(ao, WpT, bproj, (float*)d_out,
                                                  32768, 768, 768);
}
